// DGCNNEncoder_36893769072804
// MI455X (gfx1250) — matrix operand order repaired and checked by reading, not compiled, not run on hardware
//
#include <hip/hip_runtime.h>
#include <hip/hip_bf16.h>

// ---------------------------------------------------------------------------
// DGCNN encoder for MI455X (gfx1250, wave32, WMMA + Tensor Data Mover).
//  * All GEMM-shaped work -> v_wmma_f32_16x16x32_f16 (distance matrices,
//    edge MLPs, final linear).
//  * Dense 2D LDS staging (weights, linear activations, kNN feature tiles)
//    -> TDM tensor_load_to_lds (TENSORcnt), double-buffered in the kNN loop.
//    Features/weights are pre-converted once per launch to f16, padded to the
//    WMMA K-stride and weights pre-transposed, so staging is raw 2D DMA.
//  * BatchNorm batch-stats from per-channel sum/sumsq; max-over-k commutes
//    with the monotone BN+LeakyReLU (direction = sign(gamma)), so only
//    per-node max/min of pre-BN activations are kept.
// ---------------------------------------------------------------------------

#define BATCH 4
#define CIN0  6
#define NPTS  4096
#define KNN   20
#define EMB   128

typedef __attribute__((ext_vector_type(16))) _Float16 v16h;
typedef __attribute__((ext_vector_type(8)))  _Float16 v8h;
typedef __attribute__((ext_vector_type(8)))  float    v8f;
typedef __attribute__((ext_vector_type(4)))  unsigned u32x4;
typedef __attribute__((ext_vector_type(8)))  int      i32x8;
typedef __attribute__((ext_vector_type(4)))  int      i32x4;

#if defined(__HIP_DEVICE_COMPILE__) && \
    __has_builtin(__builtin_amdgcn_tensor_load_to_lds) && \
    __has_builtin(__builtin_amdgcn_s_wait_tensorcnt)
#define HAS_TDM 1
#else
#define HAS_TDM 0
#endif

// ---- TDM: async 2D tile (dim1 rows x dim0 f16 elements, row stride = dim0)
// global -> LDS.  D# packing per CDNA5 ISA ch.8 (count=1, type=2, data_size=2B).
__device__ __forceinline__ void tdm_load_2d(unsigned ldsOff, const void* gptr,
                                            int dim0, int dim1) {
#if HAS_TDM
  unsigned long long ga = (unsigned long long)(uintptr_t)gptr;
  u32x4 g0;
  g0[0] = 1u;                                            // count=1, user desc
  g0[1] = ldsOff;                                        // lds_addr (bytes)
  g0[2] = (unsigned)(ga & 0xffffffffu);                  // global_addr[31:0]
  g0[3] = (unsigned)((ga >> 32) & 0x1ffffffu) | (2u << 30); // [56:32] | type=2
  i32x8 g1;
  g1[0] = (1 << 16);                                     // data_size=1 (2B)
  g1[1] = (dim0 & 0xffff) << 16;                         // tensor_dim0 lo16
  g1[2] = ((unsigned)dim0 >> 16) | ((unsigned)(dim1 & 0xffff) << 16); // d0 hi|d1 lo
  g1[3] = ((unsigned)dim1 >> 16) | ((unsigned)dim0 << 16);            // d1 hi|tile_dim0
  g1[4] = (dim1 & 0xffff);                               // tile_dim1 (tile_dim2=0)
  g1[5] = dim0;                                          // tensor_dim0_stride lo32
  g1[6] = 0;
  g1[7] = 0;
  i32x4 z4 = {0, 0, 0, 0};
#if defined(__clang_major__) && (__clang_major__ >= 23)
  i32x8 z8 = {0, 0, 0, 0, 0, 0, 0, 0};
  __builtin_amdgcn_tensor_load_to_lds(g0, g1, z4, z4, z8, 0);
#else
  __builtin_amdgcn_tensor_load_to_lds(g0, g1, z4, z4, 0);
#endif
#else
  (void)ldsOff; (void)gptr; (void)dim0; (void)dim1;
#endif
}

__device__ __forceinline__ void tdm_wait0() {
#if HAS_TDM
  __builtin_amdgcn_s_wait_tensorcnt(0);
#endif
}

__device__ __forceinline__ unsigned lds_off(const void* p) {
  return (unsigned)(uintptr_t)p;   // LDS aperture: addr[31:0] = LDS byte offset
}

// A-fragment (16x32 f16) from LDS row-major tile (contiguous k = 16*half + s, matches B).
__device__ __forceinline__ v16h load_a_frag(const _Float16* rowPtr, int kBase, int lane) {
  const int khalf = (lane >> 4) & 1;
  return *(const v16h*)(rowPtr + kBase + khalf * 16);
}

// B-fragment (32x16 f16) from an LDS [col][k] tile: one contiguous 32B load.
__device__ __forceinline__ v16h load_b_frag(const _Float16* colKmajor, int rowStride,
                                            int nBase, int kBase, int lane) {
  const int col = nBase + (lane & 15);
  const int kOff = kBase + ((lane >> 4) & 1) * 16;
  return *(const v16h*)(colKmajor + (size_t)col * rowStride + kOff);
}

__device__ __forceinline__ v8f wmma_f16(v16h a, v16h b, v8f c) {
  return __builtin_amdgcn_wmma_f32_16x16x32_f16(false, a, false, b, (short)0, c, false, false);
}

// ---------------------------------------------------------------------------
// 0) init: zero stats + encoded global-max
// ---------------------------------------------------------------------------
__global__ void init_kernel(float* gstats, unsigned* glob) {
  int t = blockIdx.x * blockDim.x + threadIdx.x;
  if (t < 4 * 256) gstats[t] = 0.0f;
  if (t < BATCH * EMB) glob[t] = 0u;
}

// ---------------------------------------------------------------------------
// 1) prep: weights -> f16, transposed [co][k], zero-padded to KD
// ---------------------------------------------------------------------------
__global__ void prep_w_kernel(const float* __restrict__ W, _Float16* __restrict__ wt,
                              int KIN, int KD, int COUT) {
  int t = blockIdx.x * blockDim.x + threadIdx.x;
  if (t >= KD * COUT) return;
  int co = t / KD, k = t % KD;
  wt[t] = (k < KIN) ? (_Float16)W[k * COUT + co] : (_Float16)0.0f;
}

// ---------------------------------------------------------------------------
// 2) transpose x (B,C,N) -> f0h (B,N,32) f16, zero-padded
// ---------------------------------------------------------------------------
__global__ void transpose_kernel(const float* __restrict__ x, _Float16* __restrict__ f0h) {
  int t = blockIdx.x * blockDim.x + threadIdx.x;      // over B*N*32
  if (t >= BATCH * NPTS * 32) return;
  int c = t & 31;
  int n = (t >> 5) & (NPTS - 1);
  int b = t >> 17;                                    // N*32 = 2^17
  float v = (c < CIN0) ? x[((size_t)(b * CIN0 + c)) * NPTS + n] : 0.0f;
  f0h[t] = (_Float16)v;
}

// ---------------------------------------------------------------------------
// 3) kNN: TDM double-buffered column tiles, WMMA distance tiles, top-20/row
// ---------------------------------------------------------------------------
template <int CPAD>
__global__ void knn_kernel(const _Float16* __restrict__ Fh, int* __restrict__ idxOut) {
  constexpr int RB = 64, CB = 64;
  __shared__ __align__(32) _Float16 rowF[RB][CPAD];
  __shared__ __align__(32) _Float16 colF[2][CB][CPAD];
  __shared__ float rowN2[RB];
  __shared__ float colN2[CB];
  __shared__ float dT[RB][CB];

  const int tid = threadIdx.x, lane = tid & 31, wave = tid >> 5;
  const int b = blockIdx.y, rowBase = blockIdx.x * RB;
  const _Float16* Fb = Fh + (size_t)b * NPTS * CPAD;

#if HAS_TDM
  if (wave == 0) {
    tdm_load_2d(lds_off(&rowF[0][0]), Fb + (size_t)rowBase * CPAD, CPAD, RB);
    tdm_load_2d(lds_off(&colF[0][0][0]), Fb, CPAD, CB);
  }
#else
  for (int i = tid; i < RB * CPAD; i += 256)
    rowF[i / CPAD][i % CPAD] = Fb[(size_t)rowBase * CPAD + i];
#endif

  float bd[KNN]; int bi[KNN];
  for (int i = 0; i < KNN; ++i) { bd[i] = 3.4e38f; bi[i] = 0; }

  for (int cb = 0; cb < NPTS / CB; ++cb) {
    const int cur = cb & 1;
    const int colBase = cb * CB;
#if HAS_TDM
    if (wave == 0) tdm_wait0();                 // current tile (+rowF on cb=0) landed
    __syncthreads();
    if (wave == 0 && cb + 1 < NPTS / CB)        // prefetch next tile into other buffer
      tdm_load_2d(lds_off(&colF[cur ^ 1][0][0]),
                  Fb + (size_t)(colBase + CB) * CPAD, CPAD, CB);
#else
    for (int i = tid; i < CB * CPAD; i += 256)
      colF[cur][i / CPAD][i % CPAD] = Fb[(size_t)colBase * CPAD + i];
    __syncthreads();
#endif
    if (tid < CB) {                             // squared norms of column points
      float s = 0.0f;
      for (int c = 0; c < CPAD; ++c) { float v = (float)colF[cur][tid][c]; s += v * v; }
      colN2[tid] = s;
    }
    if (cb == 0 && tid >= 64 && tid < 64 + RB) {
      int r = tid - 64;
      float s = 0.0f;
      for (int c = 0; c < CPAD; ++c) { float v = (float)rowF[r][c]; s += v * v; }
      rowN2[r] = s;
    }
    __syncthreads();

    // 4 mtiles x 4 ntiles = 16 WMMA tiles -> 2 per wave (wave-uniform)
    for (int t = wave; t < 16; t += 8) {
      const int mt = t >> 2, nt = t & 3;
      v8f acc = {};
#pragma unroll
      for (int kb = 0; kb < CPAD; kb += 32) {
        v16h a = load_a_frag(&rowF[mt * 16 + (lane & 15)][0], kb, lane);
        v16h bf = load_b_frag(&colF[cur][0][0], CPAD, nt * 16, kb, lane);
        acc = wmma_f16(a, bf, acc);
      }
      const int hi = (lane >> 4) & 1;
#pragma unroll
      for (int v = 0; v < 8; ++v) {
        int m = mt * 16 + v + hi * 8;
        int n = nt * 16 + (lane & 15);
        dT[m][n] = rowN2[m] + colN2[n] - 2.0f * acc[v];
      }
    }
    __syncthreads();

    if (tid < RB) {                             // per-row top-20 insertion
      for (int c = 0; c < CB; ++c) {
        float d = dT[tid][c];
        if (d < bd[KNN - 1]) {
          int p = KNN - 1;
          while (p > 0 && bd[p - 1] > d) { bd[p] = bd[p - 1]; bi[p] = bi[p - 1]; --p; }
          bd[p] = d; bi[p] = colBase + c;
        }
      }
    }
    __syncthreads();
  }

  if (tid < RB) {
    size_t base = ((size_t)(b * NPTS + rowBase + tid)) * KNN;
    for (int i = 0; i < KNN; ++i) idxOut[base + i] = bi[i];
  }
}

// ---------------------------------------------------------------------------
// 4) EdgeConv: TDM weight tile overlapped with edge-feature gather, WMMA GEMM,
//    per-node max/min + channel stats.  4 nodes/WG -> 80 rows = 5 M-tiles.
// ---------------------------------------------------------------------------
template <int CIN, int FSTR, int KD, int COUT>
__global__ void edgeconv_kernel(const _Float16* __restrict__ Fh, const int* __restrict__ idx,
                                const _Float16* __restrict__ wtg, const float* __restrict__ bias,
                                float* __restrict__ hmax, float* __restrict__ hmin,
                                float* __restrict__ gstats) {
  constexpr int NODES = 4, ROWS = NODES * KNN;  // 80
  __shared__ __align__(32) _Float16 eT[ROWS][KD];
  __shared__ __align__(32) _Float16 Wt[COUT][KD];   // [out-col][k]
  __shared__ float hT[ROWS][16];
  __shared__ float bL[COUT];
  __shared__ float sSum[COUT], sSum2[COUT];

  const int tid = threadIdx.x, lane = tid & 31, wave = tid >> 5;
  const int b = blockIdx.y, nodeBase = blockIdx.x * NODES;
  const _Float16* Fb = Fh + (size_t)b * NPTS * FSTR;

#if HAS_TDM
  if (wave == 0) tdm_load_2d(lds_off(&Wt[0][0]), wtg, KD, COUT);  // async, overlaps gather
#else
  for (int i = tid; i < COUT * KD; i += 256) (&Wt[0][0])[i] = wtg[i];
#endif
  if (tid < COUT) { sSum[tid] = 0.0f; sSum2[tid] = 0.0f; bL[tid] = bias[tid]; }

  for (int i = tid; i < ROWS * KD; i += 256) {   // gather [xi, xj-xi], pad to KD
    int r = i / KD, c = i % KD;
    int node = nodeBase + r / KNN;
    int kk = r % KNN;
    int j = idx[((size_t)(b * NPTS + node)) * KNN + kk];
    float v = 0.0f;
    if (c < CIN) {
      v = (float)Fb[(size_t)node * FSTR + c];
    } else if (c < 2 * CIN) {
      int cc = c - CIN;
      v = (float)Fb[(size_t)j * FSTR + cc] - (float)Fb[(size_t)node * FSTR + cc];
    }
    eT[r][c] = (_Float16)v;
  }
#if HAS_TDM
  if (wave == 0) tdm_wait0();
#endif
  __syncthreads();

  for (int nt = 0; nt < COUT / 16; ++nt) {
    if (wave < 5) {                      // wave-uniform, EXEC all-1 inside
      const int mBase = wave * 16;
      v8f acc = {};
#pragma unroll
      for (int kb = 0; kb < KD; kb += 32) {
        v16h a = load_a_frag(&eT[mBase + (lane & 15)][0], kb, lane);
        v16h bf = load_b_frag(&Wt[0][0], KD, nt * 16, kb, lane);
        acc = wmma_f16(a, bf, acc);
      }
      const int hi = (lane >> 4) & 1;
#pragma unroll
      for (int v = 0; v < 8; ++v) hT[mBase + v + hi * 8][lane & 15] = acc[v];
    }
    __syncthreads();

    if (tid < NODES * 16) {
      const int node = tid >> 4, ch = tid & 15;
      const int co = nt * 16 + ch;
      const float bb = bL[co];
      float mx = -3.4e38f, mn = 3.4e38f, s = 0.0f, s2 = 0.0f;
      for (int kk = 0; kk < KNN; ++kk) {
        float v = hT[node * KNN + kk][ch] + bb;
        mx = fmaxf(mx, v); mn = fminf(mn, v); s += v; s2 += v * v;
      }
      size_t gn = (size_t)(b * NPTS + nodeBase + node);
      hmax[gn * COUT + co] = mx;
      hmin[gn * COUT + co] = mn;
      atomicAdd(&sSum[co], s);
      atomicAdd(&sSum2[co], s2);
    }
    __syncthreads();
  }

  if (tid < COUT) {
    atomicAdd(&gstats[tid], sSum[tid]);
    atomicAdd(&gstats[128 + tid], sSum2[tid]);
  }
}

// ---------------------------------------------------------------------------
// 5) finalize edge conv: BN + LReLU on max/min (sign of gamma) -> f16 features
// ---------------------------------------------------------------------------
template <int COUT>
__global__ void fin_edge_kernel(const float* __restrict__ hmax, const float* __restrict__ hmin,
                                const float* __restrict__ gstats,
                                const float* __restrict__ gamma, const float* __restrict__ beta,
                                _Float16* __restrict__ outFh, float invM) {
  int t = blockIdx.x * blockDim.x + threadIdx.x;      // over B*N*COUT
  if (t >= BATCH * NPTS * COUT) return;
  int c = t % COUT;
  float mu = gstats[c] * invM;
  float var = gstats[128 + c] * invM - mu * mu;
  float rs = rsqrtf(var + 1e-5f);
  float g = gamma[c];
  float sel = (g >= 0.0f) ? hmax[t] : hmin[t];
  float y = (sel - mu) * rs * g + beta[c];
  outFh[t] = (_Float16)((y > 0.0f) ? y : 0.2f * y);
}

// ---------------------------------------------------------------------------
// 6) final linear (B*N,128)x(128,128): TDM-staged X and W tiles, WMMA + stats
// ---------------------------------------------------------------------------
__global__ void linear_kernel(const _Float16* __restrict__ f3h, const _Float16* __restrict__ wlt,
                              const float* __restrict__ bl, float* __restrict__ h4,
                              float* __restrict__ gstats) {
  constexpr int RB = 64;
  __shared__ __align__(32) _Float16 X[RB][128];
  __shared__ __align__(32) _Float16 Wt[128][128];
  __shared__ float hT[RB][16];
  __shared__ float bL[128];
  __shared__ float sSum[128], sSum2[128];

  const int tid = threadIdx.x, lane = tid & 31, wave = tid >> 5;
  const int rowBase = blockIdx.x * RB;                // rows over B*N

#if HAS_TDM
  if (wave == 0) {
    tdm_load_2d(lds_off(&X[0][0]), f3h + (size_t)rowBase * 128, 128, RB);
    tdm_load_2d(lds_off(&Wt[0][0]), wlt, 128, 128);
  }
#else
  for (int i = tid; i < RB * 128; i += 256) (&X[0][0])[i] = f3h[(size_t)rowBase * 128 + i];
  for (int i = tid; i < 128 * 128; i += 256) (&Wt[0][0])[i] = wlt[i];
#endif
  if (tid < 128) { sSum[tid] = 0.0f; sSum2[tid] = 0.0f; bL[tid] = bl[tid]; }
#if HAS_TDM
  if (wave == 0) tdm_wait0();
#endif
  __syncthreads();

  for (int nt = 0; nt < 8; ++nt) {
    if (wave < 4) {
      const int mBase = wave * 16;
      v8f acc = {};
#pragma unroll
      for (int kb = 0; kb < 128; kb += 32) {
        v16h a = load_a_frag(&X[mBase + (lane & 15)][0], kb, lane);
        v16h bf = load_b_frag(&Wt[0][0], 128, nt * 16, kb, lane);
        acc = wmma_f16(a, bf, acc);
      }
      const int hi = (lane >> 4) & 1;
#pragma unroll
      for (int v = 0; v < 8; ++v) hT[mBase + v + hi * 8][lane & 15] = acc[v];
    }
    __syncthreads();

    if (tid < 64) {
      const int ch = tid & 15, q = tid >> 4;
      const int co = nt * 16 + ch;
      const float bb = bL[co];
      float s = 0.0f, s2 = 0.0f;
      for (int r = q * 16; r < q * 16 + 16; ++r) {
        float v = hT[r][ch] + bb;
        h4[((size_t)(rowBase + r)) * 128 + co] = v;
        s += v; s2 += v * v;
      }
      atomicAdd(&sSum[co], s);
      atomicAdd(&sSum2[co], s2);
    }
    __syncthreads();
  }

  if (tid < 128) {
    atomicAdd(&gstats[tid], sSum[tid]);
    atomicAdd(&gstats[128 + tid], sSum2[tid]);
  }
}

// ---------------------------------------------------------------------------
// 7) finalize linear: BN + LReLU -> loc; encoded-uint atomicMax global pool
// ---------------------------------------------------------------------------
__global__ void fin_lin_kernel(const float* __restrict__ h4, const float* __restrict__ gstats,
                               const float* __restrict__ gamma, const float* __restrict__ beta,
                               float* __restrict__ loc, unsigned* __restrict__ glob, float invM) {
  int t = blockIdx.x * blockDim.x + threadIdx.x;      // over B*N*128
  if (t >= BATCH * NPTS * 128) return;
  int c = t & 127;
  float mu = gstats[c] * invM;
  float var = gstats[128 + c] * invM - mu * mu;
  float rs = rsqrtf(var + 1e-5f);
  float y = (h4[t] - mu) * rs * gamma[c] + beta[c];
  y = (y > 0.0f) ? y : 0.2f * y;
  loc[t] = y;
  int b = t >> 19;                                    // N*128 = 2^19
  unsigned bits = __float_as_uint(y);
  unsigned enc = (bits & 0x80000000u) ? ~bits : (bits | 0x80000000u);
  atomicMax(&glob[(b << 7) + c], enc);
}

// ---------------------------------------------------------------------------
// 8) writeout: out (B, 2*EMB, N) = concat(loc^T, broadcast(glob))
// ---------------------------------------------------------------------------
__global__ void writeout_kernel(const float* __restrict__ loc, const unsigned* __restrict__ glob,
                                float* __restrict__ out) {
  int t = blockIdx.x * blockDim.x + threadIdx.x;      // over B*256*N
  if (t >= BATCH * 2 * EMB * NPTS) return;
  int n = t & (NPTS - 1);
  int c = (t >> 12) & 255;
  int b = t >> 20;
  float v;
  if (c < EMB) {
    v = loc[(((size_t)(b << 12)) + n) * 128 + c];
  } else {
    unsigned e = glob[(b << 7) + (c - EMB)];
    unsigned bits = (e & 0x80000000u) ? (e & 0x7fffffffu) : ~e;
    v = __uint_as_float(bits);
  }
  out[t] = v;
}

// ---------------------------------------------------------------------------
// launch
// ---------------------------------------------------------------------------
extern "C" void kernel_launch(void* const* d_in, const int* in_sizes, int n_in,
                              void* d_out, int out_size, void* d_ws, size_t ws_size,
                              hipStream_t stream) {
  (void)in_sizes; (void)n_in; (void)out_size; (void)ws_size;
  const float* x  = (const float*)d_in[0];
  const float* W1 = (const float*)d_in[1];  const float* b1 = (const float*)d_in[2];
  const float* g1 = (const float*)d_in[3];  const float* be1 = (const float*)d_in[4];
  const float* W2 = (const float*)d_in[5];  const float* b2 = (const float*)d_in[6];
  const float* g2 = (const float*)d_in[7];  const float* be2 = (const float*)d_in[8];
  const float* W3 = (const float*)d_in[9];  const float* b3 = (const float*)d_in[10];
  const float* g3 = (const float*)d_in[11]; const float* be3 = (const float*)d_in[12];
  const float* Wl = (const float*)d_in[13]; const float* bl = (const float*)d_in[14];
  const float* gl = (const float*)d_in[15]; const float* bel = (const float*)d_in[16];
  float* out = (float*)d_out;

  const size_t NB = (size_t)BATCH * NPTS;
  // byte-based workspace layout (~48 MB), 256B aligned slices
  char* p = (char*)d_ws;
  auto alloc = [&](size_t bytes) { char* r = p; p += (bytes + 255) & ~(size_t)255; return r; };
  _Float16* f0h = (_Float16*)alloc(NB * 32 * 2);
  _Float16* f1h = (_Float16*)alloc(NB * 64 * 2);
  _Float16* f2h = (_Float16*)alloc(NB * 64 * 2);
  _Float16* f3h = (_Float16*)alloc(NB * 128 * 2);
  _Float16* w1t = (_Float16*)alloc(64 * 32 * 2);
  _Float16* w2t = (_Float16*)alloc(64 * 128 * 2);
  _Float16* w3t = (_Float16*)alloc(128 * 128 * 2);
  _Float16* wlt = (_Float16*)alloc(128 * 128 * 2);
  float* h4   = (float*)alloc(NB * 128 * 4);
  float* loc  = (float*)alloc(NB * 128 * 4);
  float* hmax = (float*)alloc(NB * 128 * 4);
  float* hmin = (float*)alloc(NB * 128 * 4);
  float* gstats = (float*)alloc(4 * 256 * 4);
  unsigned* glob = (unsigned*)alloc(BATCH * EMB * 4);
  int* idx1 = (int*)alloc(NB * KNN * 4);
  int* idx2 = (int*)alloc(NB * KNN * 4);
  int* idx3 = (int*)alloc(NB * KNN * 4);

  const float invMe = 1.0f / (float)(BATCH * NPTS * KNN);
  const float invMl = 1.0f / (float)(BATCH * NPTS);

  init_kernel<<<4, 256, 0, stream>>>(gstats, glob);
  prep_w_kernel<<<(64 * 32 + 255) / 256, 256, 0, stream>>>(W1, w1t, 12, 32, 64);
  prep_w_kernel<<<(64 * 128 + 255) / 256, 256, 0, stream>>>(W2, w2t, 128, 128, 64);
  prep_w_kernel<<<(128 * 128 + 255) / 256, 256, 0, stream>>>(W3, w3t, 128, 128, 128);
  prep_w_kernel<<<(128 * 128 + 255) / 256, 256, 0, stream>>>(Wl, wlt, 128, 128, 128);
  transpose_kernel<<<(int)(NB * 32 / 256), 256, 0, stream>>>(x, f0h);

  // --- EdgeConv 1: C=6 -> 64 ---
  knn_kernel<32><<<dim3(NPTS / 64, BATCH), 256, 0, stream>>>(f0h, idx1);
  edgeconv_kernel<6, 32, 32, 64><<<dim3(NPTS / 4, BATCH), 256, 0, stream>>>(
      f0h, idx1, w1t, b1, hmax, hmin, gstats + 0 * 256);
  fin_edge_kernel<64><<<(int)(NB * 64 / 256), 256, 0, stream>>>(
      hmax, hmin, gstats + 0 * 256, g1, be1, f1h, invMe);

  // --- EdgeConv 2: C=64 -> 64 ---
  knn_kernel<64><<<dim3(NPTS / 64, BATCH), 256, 0, stream>>>(f1h, idx2);
  edgeconv_kernel<64, 64, 128, 64><<<dim3(NPTS / 4, BATCH), 256, 0, stream>>>(
      f1h, idx2, w2t, b2, hmax, hmin, gstats + 1 * 256);
  fin_edge_kernel<64><<<(int)(NB * 64 / 256), 256, 0, stream>>>(
      hmax, hmin, gstats + 1 * 256, g2, be2, f2h, invMe);

  // --- EdgeConv 3: C=64 -> 128 ---
  knn_kernel<64><<<dim3(NPTS / 64, BATCH), 256, 0, stream>>>(f2h, idx3);
  edgeconv_kernel<64, 64, 128, 128><<<dim3(NPTS / 4, BATCH), 256, 0, stream>>>(
      f2h, idx3, w3t, b3, hmax, hmin, gstats + 2 * 256);
  fin_edge_kernel<128><<<(int)(NB * 128 / 256), 256, 0, stream>>>(
      hmax, hmin, gstats + 2 * 256, g3, be3, f3h, invMe);

  // --- final linear + BN + LReLU + global max + writeout ---
  linear_kernel<<<(int)(NB / 64), 256, 0, stream>>>(f3h, wlt, bl, h4, gstats + 3 * 256);
  fin_lin_kernel<<<(int)(NB * 128 / 256), 256, 0, stream>>>(
      h4, gstats + 3 * 256, gl, bel, loc, glob, invMl);
  writeout_kernel<<<(int)(NB * 2 * EMB / 256), 256, 0, stream>>>(loc, glob, out);
}
